// HPFNetCovPool_30743375905525
// MI455X (gfx1250) — compile-verified, matrix operand order checked by reading, not run on hardware
//
#include <hip/hip_runtime.h>

// ---------------------------------------------------------------------------
// MI455X (gfx1250) implementation of HPFNet cov-pool head:
//   x = w_bn @ features (1x1 conv, bias cancels under centering)
//   cov = centered(x) @ centered(x)^T / 256
//   s = NewtonSchulz_sqrtm(cov, 5) ; vec = triu(s) ; two linear heads
// All heavy math in bf16 WMMA (v_wmma_f32_16x16x32_bf16), f32 accumulate.
// ---------------------------------------------------------------------------

typedef unsigned short u16;
typedef __attribute__((ext_vector_type(16))) __bf16 v16bf;
typedef __attribute__((ext_vector_type(8)))  float  v8f;

union FragBits { uint4 u[2]; v16bf v; };

__device__ __forceinline__ u16 f2bf(float f) {
  unsigned int b = __float_as_uint(f);
  return (u16)((b + 0x7FFFu + ((b >> 16) & 1u)) >> 16);   // RNE
}

// Load one 16x32 bf16 WMMA operand fragment from an LDS-resident row-major
// tile.  base_row = tile's first row (for A: M-tile row; for B of a symmetric
// matrix: N-tile col, reading rows as columns).  ISA 7.12.2 16-bit layout:
//   lanes 0-15 : row base+lane,  K = k0+{0..7}  then k0+{16..23}
//   lanes 16-31: row base+lane-16, K = k0+{8..15} then k0+{24..31}
__device__ __forceinline__ v16bf load_frag(const u16* lds, int base_row,
                                           int k0, int pitch) {
  int lane = threadIdx.x & 31;
  int row  = base_row + (lane & 15);
  int kh   = (lane >> 4) << 3;                 // 0 or 8
  const u16* p = lds + row * pitch + k0 + kh;
  FragBits f;
  f.u[0] = *(const uint4*)(p);
  f.u[1] = *(const uint4*)(p + 16);
  return f.v;
}

// ---------------------------------------------------------------------------
// Prep: f32 -> bf16 weight convert
// ---------------------------------------------------------------------------
__global__ void k_cvt_wbn(const float* __restrict__ w, u16* __restrict__ o, int n) {
  int i = blockIdx.x * 256 + threadIdx.x;
  if (i < n) o[i] = f2bf(w[i]);
}

// Prep: features [B,2048,256] f32 -> Ft [B,256,2048] bf16 (tiled transpose)
__global__ void k_transpose_feat(const float* __restrict__ F, u16* __restrict__ Ft) {
  __shared__ float tile[32][33];
  int bid = blockIdx.x;
  int nb = bid & 7;                 // 256/32 spatial tiles
  int cb = (bid >> 3) & 63;         // 2048/32 channel tiles
  int b  = bid >> 9;
  int tx = threadIdx.x & 31, ty = threadIdx.x >> 5;   // 32 x 8
  const float* Fb = F + ((size_t)b * 2048 + (size_t)cb * 32) * 256 + nb * 32;
  #pragma unroll
  for (int i = 0; i < 32; i += 8)
    tile[ty + i][tx] = Fb[(size_t)(ty + i) * 256 + tx];
  __syncthreads();
  u16* Fo = Ft + ((size_t)b * 256 + (size_t)nb * 32) * 2048 + cb * 32;
  #pragma unroll
  for (int i = 0; i < 32; i += 8)
    Fo[(size_t)(ty + i) * 2048 + tx] = f2bf(tile[tx][ty + i]);
}

// ---------------------------------------------------------------------------
// Conv GEMM + centering:  Xc[b] = center_rows( Wbn(256x2048) @ F[b](2048x256) )
// grid = B*4 (64-row stripes), block = 256 (8 waves).
// Wave w computes rows 16*(w&3) of stripe, cols 128*(w>>2)..+128 (8 tiles).
// ---------------------------------------------------------------------------
#define CKT 64
#define CP  72          // LDS pitch (bf16 elems) for staging
#define XP  260         // LDS pitch (f32) for X stripe

__global__ void k_conv_center(const u16* __restrict__ Wbn,
                              const u16* __restrict__ Ft,
                              u16* __restrict__ Xc) {
  extern __shared__ __align__(16) char smem[];
  u16* As = (u16*)smem;             // 64 x CP
  u16* Bs = As + 64 * CP;           // 256 x CP
  int tid = threadIdx.x;
  int b = blockIdx.x >> 2;
  int stripe = (blockIdx.x & 3) * 64;
  const u16* Wb = Wbn + (size_t)stripe * 2048;
  const u16* Fb = Ft + (size_t)b * 256 * 2048;

  int wave = tid >> 5;
  int rt = (wave & 3) * 16;         // row-tile within stripe
  int c0 = (wave >> 2) * 128;       // column half

  v8f acc[8];
  #pragma unroll
  for (int t = 0; t < 8; t++) acc[t] = (v8f){0.f,0.f,0.f,0.f,0.f,0.f,0.f,0.f};

  for (int kk = 0; kk < 2048; kk += CKT) {
    __syncthreads();
    for (int i = tid; i < 64 * (CKT / 8); i += 256) {
      int r = i >> 3, c8 = (i & 7) * 8;
      *(uint4*)(As + r * CP + c8) = *(const uint4*)(Wb + (size_t)r * 2048 + kk + c8);
    }
    for (int i = tid; i < 256 * (CKT / 8); i += 256) {
      int r = i >> 3, c8 = (i & 7) * 8;
      *(uint4*)(Bs + r * CP + c8) = *(const uint4*)(Fb + (size_t)r * 2048 + kk + c8);
      if (kk + CKT < 2048)
        __builtin_prefetch(Fb + (size_t)r * 2048 + kk + CKT + c8, 0, 3);
    }
    __syncthreads();
    #pragma unroll
    for (int ks = 0; ks < CKT; ks += 32) {
      v16bf a = load_frag(As, rt, ks, CP);
      #pragma unroll
      for (int t = 0; t < 8; t++) {
        v16bf bf = load_frag(Bs, c0 + t * 16, ks, CP);
        acc[t] = __builtin_amdgcn_wmma_f32_16x16x32_bf16(
            false, a, false, bf, (short)0, acc[t], false, false);
      }
    }
  }
  __syncthreads();

  // spill stripe to LDS (f32), reduce row means, emit centered bf16
  float* Xs  = (float*)smem;              // 64 x XP
  float* red = Xs + 64 * XP;              // 256 partials
  float* mu  = red + 256;                 // 64 means
  int lane = tid & 31, n = lane & 15, mb8 = (lane >> 4) * 8;
  #pragma unroll
  for (int t = 0; t < 8; t++)
    #pragma unroll
    for (int r = 0; r < 8; r++)
      Xs[(rt + mb8 + r) * XP + c0 + t * 16 + n] = acc[t][r];
  __syncthreads();

  int rr = tid >> 2, q = tid & 3;
  float s = 0.f;
  for (int c = q * 64; c < q * 64 + 64; c++) s += Xs[rr * XP + c];
  red[tid] = s;
  __syncthreads();
  if (q == 0)
    mu[rr] = (red[tid] + red[tid + 1] + red[tid + 2] + red[tid + 3]) * (1.f / 256.f);
  __syncthreads();

  u16* Xo = Xc + (size_t)b * 65536 + (size_t)stripe * 256;
  for (int i = tid; i < 64 * 256; i += 256) {
    int r = i >> 8, c = i & 255;
    Xo[i] = f2bf(Xs[r * XP + c] - mu[r]);
  }
}

// ---------------------------------------------------------------------------
// Generic batched 256x256x256 bf16 GEMM, B-operand read via symmetry
// (column n of B == row n).  grid = B*2 (128-row stripes), block = 256.
// mode 0: D(bf16) = acc
// mode 1: D(bf16) = 0.5*(3I - acc)            (Newton-Schulz polynomial)
// mode 2: D(f32)  = acc * scales[batch]       (final *sqrt(trace))
// mode 3: D(f32)  = acc * (1/256)             (covariance /M)
// ---------------------------------------------------------------------------
#define GP 264          // LDS pitch (bf16 elems)

__global__ void k_ns_gemm(const u16* __restrict__ A, const u16* __restrict__ B,
                          void* __restrict__ D, const float* __restrict__ scales,
                          int mode) {
  extern __shared__ __align__(16) char smem[];
  u16* As = (u16*)smem;             // 128 x GP
  u16* Bs = As + 128 * GP;          // 256 x GP
  int tid = threadIdx.x;
  int b = blockIdx.x >> 1;
  int stripe = (blockIdx.x & 1) * 128;
  const u16* Ab = A + (size_t)b * 65536 + (size_t)stripe * 256;
  const u16* Bb = B + (size_t)b * 65536;

  for (int i = tid; i < 128 * 32; i += 256) {
    int r = i >> 5, c8 = (i & 31) * 8;
    *(uint4*)(As + r * GP + c8) = *(const uint4*)(Ab + (size_t)r * 256 + c8);
  }
  for (int i = tid; i < 256 * 32; i += 256) {
    int r = i >> 5, c8 = (i & 31) * 8;
    *(uint4*)(Bs + r * GP + c8) = *(const uint4*)(Bb + (size_t)r * 256 + c8);
  }
  __syncthreads();

  int wave = tid >> 5;
  int row0 = wave * 16;             // wave owns one 16-row tile, 16 col tiles
  v8f acc[16];
  #pragma unroll
  for (int t = 0; t < 16; t++) acc[t] = (v8f){0.f,0.f,0.f,0.f,0.f,0.f,0.f,0.f};

  #pragma unroll 2
  for (int k0 = 0; k0 < 256; k0 += 32) {
    v16bf a = load_frag(As, row0, k0, GP);
    #pragma unroll
    for (int t = 0; t < 16; t++) {
      v16bf bf = load_frag(Bs, t * 16, k0, GP);
      acc[t] = __builtin_amdgcn_wmma_f32_16x16x32_bf16(
          false, a, false, bf, (short)0, acc[t], false, false);
    }
  }

  int lane = tid & 31, n = lane & 15, mb8 = (lane >> 4) * 8;
  size_t obase = (size_t)b * 65536;
  float scl = (mode == 2) ? scales[b] : (1.f / 256.f);
  #pragma unroll
  for (int t = 0; t < 16; t++) {
    int gcol = t * 16 + n;
    #pragma unroll
    for (int r = 0; r < 8; r++) {
      int grow = stripe + row0 + mb8 + r;
      float v = acc[t][r];
      size_t idx = obase + (size_t)grow * 256 + gcol;
      if (mode == 0) {
        ((u16*)D)[idx] = f2bf(v);
      } else if (mode == 1) {
        ((u16*)D)[idx] = f2bf(0.5f * ((grow == gcol ? 3.f : 0.f) - v));
      } else {
        ((float*)D)[idx] = v * scl;
      }
    }
  }
}

// ---------------------------------------------------------------------------
// NS init: trace(cov) per batch; An = cov/tr (bf16), ZY0 = 0.5(3I-An) (bf16),
// sqrtnorm[b] = sqrt(tr).  grid = B, block = 256.
// ---------------------------------------------------------------------------
__global__ void k_ns_init(const float* __restrict__ Cov, u16* __restrict__ An,
                          u16* __restrict__ Z0, float* __restrict__ sqrtnorm) {
  __shared__ float red[256];
  __shared__ float trsh;
  int b = blockIdx.x, tid = threadIdx.x;
  const float* Cb = Cov + (size_t)b * 65536;
  red[tid] = Cb[(size_t)tid * 256 + tid];
  __syncthreads();
  for (int s = 128; s > 0; s >>= 1) {
    if (tid < s) red[tid] += red[tid + s];
    __syncthreads();
  }
  if (tid == 0) { trsh = red[0]; sqrtnorm[b] = sqrtf(red[0]); }
  __syncthreads();
  float inv = 1.f / trsh;
  u16* Ao = An + (size_t)b * 65536;
  u16* Zo = Z0 + (size_t)b * 65536;
  for (int i = tid; i < 65536; i += 256) {
    int r = i >> 8, c = i & 255;
    float an = Cb[i] * inv;
    Ao[i] = f2bf(an);
    Zo[i] = f2bf(0.5f * ((r == c ? 3.f : 0.f) - an));
  }
}

// ---------------------------------------------------------------------------
// Heads: triu-vec of S then 4-class + 1-flag linear heads. grid = B.
// ---------------------------------------------------------------------------
__global__ void k_heads(const float* __restrict__ S,
                        const float* __restrict__ wt, const float* __restrict__ bt,
                        const float* __restrict__ wf, const float* __restrict__ bfl,
                        float* __restrict__ out) {
  __shared__ float red[256];
  int b = blockIdx.x, tid = threadIdx.x;
  const float* Sb = S + (size_t)b * 65536;
  float a0 = 0.f, a1 = 0.f, a2 = 0.f, a3 = 0.f, a4 = 0.f;
  for (int r = 0; r < 256; r++) {
    int base = r * 256 - (r * (r - 1)) / 2;     // triu row offset
    const float* srow = Sb + (size_t)r * 256;
    for (int c = r + tid; c < 256; c += 256) {
      float s = srow[c];
      int w = base + (c - r);
      a0 += s * wt[w];
      a1 += s * wt[32896 + w];
      a2 += s * wt[2 * 32896 + w];
      a3 += s * wt[3 * 32896 + w];
      a4 += s * wf[w];
    }
  }
  float accs[5] = {a0, a1, a2, a3, a4};
  float result[5];
  for (int o = 0; o < 5; o++) {
    red[tid] = accs[o];
    __syncthreads();
    for (int s = 128; s > 0; s >>= 1) {
      if (tid < s) red[tid] += red[tid + s];
      __syncthreads();
    }
    if (tid == 0) result[o] = red[0];
    __syncthreads();
  }
  if (tid == 0) {
    out[b * 4 + 0] = result[0] + bt[0];
    out[b * 4 + 1] = result[1] + bt[1];
    out[b * 4 + 2] = result[2] + bt[2];
    out[b * 4 + 3] = result[3] + bt[3];
    out[256 + b]   = result[4] + bfl[0];
  }
}

// ---------------------------------------------------------------------------
extern "C" void kernel_launch(void* const* d_in, const int* in_sizes, int n_in,
                              void* d_out, int out_size, void* d_ws, size_t ws_size,
                              hipStream_t stream) {
  const float* features = (const float*)d_in[0];   // [64,2048,16,16]
  const float* w_bn     = (const float*)d_in[1];   // [256,2048]
  // d_in[2] = b_bn: cancels under centering (and is zeros) -> unused
  const float* w_type   = (const float*)d_in[3];   // [4,32896]
  const float* b_type   = (const float*)d_in[4];
  const float* w_flag   = (const float*)d_in[5];   // [1,32896]
  const float* b_flag   = (const float*)d_in[6];
  float* out = (float*)d_out;                      // 64*4 type ++ 64*1 flag

  // workspace carve-up (all 256B aligned)
  char* ws = (char*)d_ws;
  size_t off = 0;
  auto take = [&](size_t bytes) { char* p = ws + off; off = (off + bytes + 255) & ~(size_t)255; return p; };
  u16*   Wbf  = (u16*)take((size_t)256 * 2048 * 2);          // 1 MB
  u16*   Ft   = (u16*)take((size_t)64 * 256 * 2048 * 2);     // 64 MB
  u16*   Xc   = (u16*)take((size_t)64 * 65536 * 2);          // 8 MB
  u16*   Ya   = (u16*)take((size_t)64 * 65536 * 2);
  u16*   Yb   = (u16*)take((size_t)64 * 65536 * 2);
  u16*   Za   = (u16*)take((size_t)64 * 65536 * 2);
  u16*   Zb   = (u16*)take((size_t)64 * 65536 * 2);
  u16*   ZY   = (u16*)take((size_t)64 * 65536 * 2);
  float* Sbuf = (float*)take((size_t)64 * 65536 * 4);        // 16 MB
  float* snrm = (float*)take((size_t)64 * 4);

  const size_t smem_conv = (size_t)(64 * XP + 256 + 64) * 4 + 256;       // ~68 KB
  const size_t smem_gemm = (size_t)(128 + 256) * GP * sizeof(u16);       // ~203 KB

  // 1) weight convert + feature transpose/convert
  k_cvt_wbn<<<2048, 256, 0, stream>>>(w_bn, Wbf, 256 * 2048);
  k_transpose_feat<<<32768, 256, 0, stream>>>(features, Ft);

  // 2) conv GEMM + centering -> Xc (bf16)
  k_conv_center<<<256, 256, smem_conv, stream>>>(Wbf, Ft, Xc);

  // 3) cov = Xc @ Xc^T / 256 -> Sbuf (f32)   [mode 3]
  k_ns_gemm<<<128, 256, smem_gemm, stream>>>(Xc, Xc, Sbuf, nullptr, 3);

  // 4) trace-normalize: An->Ya, ZY0->Za, sqrt(tr)->snrm
  k_ns_init<<<64, 256, 0, stream>>>(Sbuf, Ya, Za, snrm);

  // 5) Y1 = An @ ZY0 -> Yb   (Z stays = ZY0 in Za)
  k_ns_gemm<<<128, 256, smem_gemm, stream>>>(Ya, Za, Yb, nullptr, 0);

  u16 *Ycur = Yb, *Ysp = Ya, *Zcur = Za, *Zsp = Zb;
  for (int it = 0; it < 3; it++) {   // ITER_N-2
    k_ns_gemm<<<128, 256, smem_gemm, stream>>>(Zcur, Ycur, ZY, nullptr, 1);  // ZY=0.5(3I-Z@Y)
    k_ns_gemm<<<128, 256, smem_gemm, stream>>>(Ycur, ZY, Ysp, nullptr, 0);   // Y = Y@ZY
    k_ns_gemm<<<128, 256, smem_gemm, stream>>>(ZY, Zcur, Zsp, nullptr, 0);   // Z = ZY@Z
    u16* t;
    t = Ycur; Ycur = Ysp; Ysp = t;
    t = Zcur; Zcur = Zsp; Zsp = t;
  }

  // 6) final: ZY = 0.5(3I - Z@Y);  S = (Y @ ZY) * sqrt(trace)  (f32)
  k_ns_gemm<<<128, 256, smem_gemm, stream>>>(Zcur, Ycur, ZY, nullptr, 1);
  k_ns_gemm<<<128, 256, smem_gemm, stream>>>(Ycur, ZY, Sbuf, snrm, 2);

  // 7) triu-vec + linear heads
  k_heads<<<64, 256, 0, stream>>>(Sbuf, w_type, b_type, w_flag, b_flag, out);
}
